// NeuralFingerprint_78125455114338
// MI455X (gfx1250) — matrix operand order repaired and checked by reading, not compiled, not run on hardware
//
#include <hip/hip_runtime.h>
#include <hip/hip_bf16.h>

typedef __attribute__((ext_vector_type(16))) _Float16 v16h;
typedef __attribute__((ext_vector_type(8)))  _Float16 v8h;
typedef __attribute__((ext_vector_type(8)))  float    v8f;

#define FD 64
#define LFP 128

// ---------------- embedding lookup: emb0[n][k] = table[feat[n]][k] ----------------
__global__ void nf_embed(const float* __restrict__ table, const int* __restrict__ feat,
                         float* __restrict__ emb, int N) {
    int i = blockIdx.x * blockDim.x + threadIdx.x;
    if (i >= N * FD) return;
    int n = i >> 6, k = i & 63;
    emb[i] = table[feat[n] * FD + k];
}

// ---------------- CSR build ----------------
__global__ void nf_count(const int* __restrict__ dst, int* __restrict__ deg, int E) {
    int e = blockIdx.x * blockDim.x + threadIdx.x;
    if (e < E) atomicAdd(&deg[dst[e]], 1);
}

__global__ void nf_scanA(const int* __restrict__ deg, int* __restrict__ excl,
                         int* __restrict__ bsum, int n) {
    __shared__ int sm[256];
    int i = blockIdx.x * 256 + threadIdx.x;
    int x = (i < n) ? deg[i] : 0;
    sm[threadIdx.x] = x;
    __syncthreads();
    for (int off = 1; off < 256; off <<= 1) {
        int t = (threadIdx.x >= off) ? sm[threadIdx.x - off] : 0;
        __syncthreads();
        sm[threadIdx.x] += t;
        __syncthreads();
    }
    if (i < n) excl[i] = sm[threadIdx.x] - x;
    if (threadIdx.x == 255) bsum[blockIdx.x] = sm[255];
}

__global__ void nf_scanB(const int* __restrict__ bsum, int* __restrict__ boff, int nb) {
    __shared__ int sm[256];
    int x = (threadIdx.x < nb) ? bsum[threadIdx.x] : 0;
    sm[threadIdx.x] = x;
    __syncthreads();
    for (int off = 1; off < 256; off <<= 1) {
        int t = (threadIdx.x >= off) ? sm[threadIdx.x - off] : 0;
        __syncthreads();
        sm[threadIdx.x] += t;
        __syncthreads();
    }
    if (threadIdx.x < nb) boff[threadIdx.x] = sm[threadIdx.x] - x;
}

__global__ void nf_scanC(const int* __restrict__ excl, const int* __restrict__ boff,
                         int* __restrict__ rowptr, int* __restrict__ cursor, int n, int E) {
    int i = blockIdx.x * blockDim.x + threadIdx.x;
    if (i < n) {
        int rp = excl[i] + boff[i >> 8];
        rowptr[i] = rp;
        cursor[i] = rp;
    }
    if (i == 0) rowptr[n] = E;
}

__global__ void nf_fill(const int* __restrict__ src, const int* __restrict__ dst,
                        int* __restrict__ cursor, int* __restrict__ col, int E) {
    int e = blockIdx.x * blockDim.x + threadIdx.x;
    if (e < E) {
        int p = atomicAdd(&cursor[dst[e]], 1);
        col[p] = src[e];
    }
}

// ---------------- gather: v[n] = emb[n] + sum_{nbr} emb[nbr]  (one wave per node) ----------------
__global__ void nf_gather(const float* __restrict__ emb, const int* __restrict__ rowptr,
                          const int* __restrict__ col, float* __restrict__ v, int N) {
    int gid  = blockIdx.x * blockDim.x + threadIdx.x;
    int node = gid >> 5;
    int lane = gid & 31;
    if (node >= N) return;
    const float2* base = (const float2*)(emb + (size_t)node * FD);
    float2 acc = base[lane];
    int e0 = rowptr[node], e1 = rowptr[node + 1];
    for (int e = e0; e < e1; ++e) {
        int s = col[e];                       // uniform across wave -> scalar load
        float2 t = *(const float2*)(emb + (size_t)s * FD + lane * 2);
        acc.x += t.x;
        acc.y += t.y;
    }
    *((float2*)(v + (size_t)node * FD) + lane) = acc;
}

// ---------------- fused per-round MLP: h=relu(v@Wh+bh); p=softmax(h@Wfp+bfp); f+=sum p ----------
// 8 waves/block, one 16-node tile per wave. WMMA f32_16x16x32_f16.
#define WPITCH 72   // padded K-pitch (halfs): 144B = 36 banks -> conflict-free fragment reads

__device__ __forceinline__ v16h nf_loadB(const _Float16* Wt, int jm, int g, int s) {
    const _Float16* p = Wt + jm * WPITCH + 8 * g + 32 * s;   // 16B-aligned
    v8h lo = *(const v8h*)p;
    v8h hi = *(const v8h*)(p + 16);
    v16h r;
#pragma unroll
    for (int i = 0; i < 8; ++i) { r[i] = lo[i]; r[8 + i] = hi[i]; }
    return r;
}

__global__ __launch_bounds__(256) void nf_mlp(
    const float* __restrict__ v, float* __restrict__ emb_out,
    const float* __restrict__ Wh, const float* __restrict__ bh,
    const float* __restrict__ Wfp, const float* __restrict__ bfp,
    float* __restrict__ fglob, int N)
{
    __shared__ _Float16 sWh[FD * WPITCH];        // transposed: [n][k]
    __shared__ _Float16 sWfp[LFP * WPITCH];      // transposed: [n][k]
    __shared__ _Float16 sH[8 * 16 * WPITCH];     // per-wave h staging [m][k]
    __shared__ float sBh[FD];
    __shared__ float sBfp[LFP];
    __shared__ float sF[LFP];

    int t = threadIdx.x;
    for (int i = t; i < FD * FD; i += 256) {
        int k = i >> 6, n = i & 63;
        sWh[n * WPITCH + k] = (_Float16)Wh[i];
    }
    for (int i = t; i < FD * LFP; i += 256) {
        int k = i >> 7, n = i & 127;
        sWfp[n * WPITCH + k] = (_Float16)Wfp[i];
    }
    if (t < FD)  sBh[t] = bh[t];
    if (t < LFP) { sBfp[t] = bfp[t]; sF[t] = 0.f; }
    __syncthreads();

    int w = t >> 5, lane = t & 31;
    int ml = lane & 15, g = lane >> 4;
    int node_base = (blockIdx.x * 8 + w) * 16;

    // --- A fragments for GEMM1 straight from v (f32 -> f16) ---
    int arow = node_base + ml;
    if (arow >= N) arow = N - 1;                 // clamp; stores are masked below
    const float* vp = v + (size_t)arow * FD;
    v16h afr[2];
#pragma unroll
    for (int s = 0; s < 2; ++s) {
        int b0 = 8 * g + 32 * s;
#pragma unroll
        for (int i = 0; i < 4; ++i) {
            float2 p0 = *(const float2*)(vp + b0 + 2 * i);
            afr[s][2 * i]     = (_Float16)p0.x;
            afr[s][2 * i + 1] = (_Float16)p0.y;
            float2 p1 = *(const float2*)(vp + b0 + 16 + 2 * i);
            afr[s][8 + 2 * i] = (_Float16)p1.x;
            afr[s][9 + 2 * i] = (_Float16)p1.y;
        }
    }

    // --- GEMM1: 16x64 = (16x64)@(64x64), 4 N-tiles x 2 K-steps ---
    v8f zero = {};
    v8f acc[4] = {zero, zero, zero, zero};
#pragma unroll
    for (int s = 0; s < 2; ++s)
#pragma unroll
        for (int j = 0; j < 4; ++j)
            acc[j] = __builtin_amdgcn_wmma_f32_16x16x32_f16(
                false, afr[s], false, nf_loadB(sWh, 16 * j + ml, g, s),
                (short)0, acc[j], false, false);

    // --- bias + ReLU; write h to staging (A-layout source) and to emb_out ---
    _Float16* hs = sH + w * (16 * WPITCH);
#pragma unroll
    for (int j = 0; j < 4; ++j) {
        int n = 16 * j + ml;
        float bb = sBh[n];
#pragma unroll
        for (int r = 0; r < 8; ++r) {
            float hv = acc[j][r] + bb;
            hv = hv > 0.f ? hv : 0.f;
            int mr = r + 8 * g;
            hs[mr * WPITCH + n] = (_Float16)hv;
            int node = node_base + mr;
            if (node < N) emb_out[(size_t)node * FD + n] = hv;
        }
    }
    __syncthreads();

    // --- GEMM2: 16x128 = (16x64)@(64x128), 8 N-tiles x 2 K-steps ---
    v8f acc2[8] = {zero, zero, zero, zero, zero, zero, zero, zero};
#pragma unroll
    for (int s = 0; s < 2; ++s) {
        const _Float16* hp = hs + ml * WPITCH + 8 * g + 32 * s;
        v8h lo = *(const v8h*)hp;
        v8h hi = *(const v8h*)(hp + 16);
        v16h a2;
#pragma unroll
        for (int i = 0; i < 8; ++i) { a2[i] = lo[i]; a2[8 + i] = hi[i]; }
#pragma unroll
        for (int j = 0; j < 8; ++j)
            acc2[j] = __builtin_amdgcn_wmma_f32_16x16x32_f16(
                false, a2, false, nf_loadB(sWfp, 16 * j + ml, g, s),
                (short)0, acc2[j], false, false);
    }

    // --- bias, per-row softmax over 128 via half-wave shuffles, fingerprint accum ---
#pragma unroll
    for (int j = 0; j < 8; ++j) {
        float bb = sBfp[16 * j + ml];
#pragma unroll
        for (int r = 0; r < 8; ++r) acc2[j][r] += bb;
    }
    float fpart[8];
#pragma unroll
    for (int j = 0; j < 8; ++j) fpart[j] = 0.f;

#pragma unroll
    for (int r = 0; r < 8; ++r) {
        float mx = acc2[0][r];
#pragma unroll
        for (int j = 1; j < 8; ++j) mx = fmaxf(mx, acc2[j][r]);
#pragma unroll
        for (int off = 1; off < 16; off <<= 1) mx = fmaxf(mx, __shfl_xor(mx, off, 16));
        float ex[8], sum = 0.f;
#pragma unroll
        for (int j = 0; j < 8; ++j) { ex[j] = __expf(acc2[j][r] - mx); sum += ex[j]; }
#pragma unroll
        for (int off = 1; off < 16; off <<= 1) sum += __shfl_xor(sum, off, 16);
        float inv = 1.f / sum;
        int node = node_base + r + 8 * g;
        if (node < N) {
#pragma unroll
            for (int j = 0; j < 8; ++j) fpart[j] += ex[j] * inv;
        }
    }
#pragma unroll
    for (int j = 0; j < 8; ++j) unsafeAtomicAdd(&sF[16 * j + ml], fpart[j]);
    __syncthreads();
    if (t < LFP) unsafeAtomicAdd(&fglob[t], sF[t]);
}

// ---------------- final classifier: softmax(f @ Wcl + bcl) -> 10 ----------------
__global__ void nf_final(const float* __restrict__ f, const float* __restrict__ Wcl,
                         const float* __restrict__ bcl, float* __restrict__ out) {
    __shared__ float lg[10];
    int c = threadIdx.x;
    if (c < 10) {
        float a = bcl[c];
        for (int k = 0; k < LFP; ++k) a += f[k] * Wcl[k * 10 + c];
        lg[c] = a;
    }
    __syncthreads();
    if (c == 0) {
        float m = lg[0];
        for (int i = 1; i < 10; ++i) m = fmaxf(m, lg[i]);
        float e[10], s = 0.f;
        for (int i = 0; i < 10; ++i) { e[i] = __expf(lg[i] - m); s += e[i]; }
        for (int i = 0; i < 10; ++i) out[i] = e[i] / s;
    }
}

extern "C" void kernel_launch(void* const* d_in, const int* in_sizes, int n_in,
                              void* d_out, int out_size, void* d_ws, size_t ws_size,
                              hipStream_t stream) {
    const int*   feat  = (const int*)d_in[0];
    const int*   esrc  = (const int*)d_in[1];
    const int*   edst  = (const int*)d_in[2];
    const float* table = (const float*)d_in[3];
    const float* Wh    = (const float*)d_in[4];
    const float* bh    = (const float*)d_in[5];
    const float* Wfp   = (const float*)d_in[6];
    const float* bfp   = (const float*)d_in[7];
    const float* Wcl   = (const float*)d_in[8];
    const float* bcl   = (const float*)d_in[9];
    const int N = in_sizes[0];
    const int E = in_sizes[1];

    char* p = (char*)d_ws;
    auto alloc = [&](size_t bytes) -> char* {
        char* r = p; p += (bytes + 255) & ~(size_t)255; return r;
    };
    float* emb0   = (float*)alloc((size_t)N * FD * 4);
    float* emb1   = (float*)alloc((size_t)N * FD * 4);
    float* vbuf   = (float*)alloc((size_t)N * FD * 4);
    float* fac    = (float*)alloc(512);
    int*   deg    = (int*)alloc((size_t)N * 4);
    int*   excl   = (int*)alloc((size_t)N * 4);
    int*   bsum   = (int*)alloc(1024);
    int*   boff   = (int*)alloc(1024);
    int*   rowptr = (int*)alloc((size_t)(N + 1) * 4);
    int*   cursor = (int*)alloc((size_t)N * 4);
    int*   col    = (int*)alloc((size_t)E * 4);

    hipMemsetAsync(deg, 0, (size_t)N * 4, stream);
    hipMemsetAsync(fac, 0, 512, stream);

    nf_embed<<<(N * FD + 255) / 256, 256, 0, stream>>>(table, feat, emb0, N);

    // CSR build (once per launch, reused for all 4 rounds)
    nf_count<<<(E + 255) / 256, 256, 0, stream>>>(edst, deg, E);
    int nb = (N + 255) / 256;
    nf_scanA<<<nb, 256, 0, stream>>>(deg, excl, bsum, N);
    nf_scanB<<<1, 256, 0, stream>>>(bsum, boff, nb);
    nf_scanC<<<nb, 256, 0, stream>>>(excl, boff, rowptr, cursor, N, E);
    nf_fill<<<(E + 255) / 256, 256, 0, stream>>>(esrc, edst, cursor, col, E);

    float* cur = emb0;
    float* nxt = emb1;
    int tiles   = (N + 15) / 16;
    int mblocks = (tiles + 7) / 8;
    for (int r = 0; r < 4; ++r) {
        nf_gather<<<((size_t)N * 32 + 255) / 256, 256, 0, stream>>>(cur, rowptr, col, vbuf, N);
        nf_mlp<<<mblocks, 256, 0, stream>>>(vbuf, nxt,
                                            Wh + (size_t)r * FD * FD, bh + (size_t)r * FD,
                                            Wfp + (size_t)r * FD * LFP, bfp + (size_t)r * LFP,
                                            fac, N);
        float* tm = cur; cur = nxt; nxt = tm;
    }
    nf_final<<<1, 32, 0, stream>>>(fac, Wcl, bcl, (float*)d_out);
}
